// Who2com_73366631350403
// MI455X (gfx1250) — matrix operand order repaired and checked by reading, not compiled, not run on hardware
//
#include <hip/hip_runtime.h>
#include <stdint.h>

// ---------------------------------------------------------------------------
// Types for CDNA5 WMMA (wave32): v_wmma_f32_16x16x32_bf16
// ---------------------------------------------------------------------------
typedef __bf16 bf16;
typedef __attribute__((ext_vector_type(16))) __bf16 v16bf;
typedef __attribute__((ext_vector_type(8)))  __bf16 bf16x8;
typedef __attribute__((ext_vector_type(8)))  float  v8f;
typedef __attribute__((ext_vector_type(4)))  int    v4i;

static __device__ __forceinline__ v8f vzero8f() {
  v8f v;
#pragma unroll
  for (int i = 0; i < 8; ++i) v[i] = 0.0f;
  return v;
}
static __device__ __forceinline__ bf16x8 bzero8() {
  bf16x8 v;
#pragma unroll
  for (int i = 0; i < 8; ++i) v[i] = (bf16)0.0f;
  return v;
}

// ---------------------------------------------------------------------------
// gfx1250 async global->LDS path for weight tiles (ASYNCcnt-tracked).
// Builtin expects (v4i AS1* src, v4i AS3* dst, imm offset, imm cpol)
// (clang prints AS1 as "__device__" in HIP mode).
// ---------------------------------------------------------------------------
#if defined(__gfx1250__) && __has_builtin(__builtin_amdgcn_global_load_async_to_lds_b128)
#define HAVE_ASYNC_LDS 1
#else
#define HAVE_ASYNC_LDS 0
#endif

#if HAVE_ASYNC_LDS
typedef __attribute__((address_space(1))) v4i gv4i;  // global
typedef __attribute__((address_space(3))) v4i lv4i;  // LDS
static __device__ __forceinline__ void async_b128(const bf16* g, bf16* l) {
  __builtin_amdgcn_global_load_async_to_lds_b128((gv4i*)g, (lv4i*)l, 0, 0);
}
static __device__ __forceinline__ void async_wait0() {
#if __has_builtin(__builtin_amdgcn_s_wait_asynccnt)
  __builtin_amdgcn_s_wait_asynccnt(0);
#else
  asm volatile("s_wait_asynccnt 0x0" ::: "memory");
#endif
}
#endif

// ---------------------------------------------------------------------------
// Pack kernels: f32 -> bf16 with layout changes
// ---------------------------------------------------------------------------
// bevs (160,256,16,16) NCHW f32 -> (160,16,16,256) NHWC bf16
__global__ void pack_bevs_k(const float* __restrict__ src, bf16* __restrict__ dst, int total) {
  for (int i = blockIdx.x * blockDim.x + threadIdx.x; i < total; i += gridDim.x * blockDim.x) {
    int c  = i & 255;
    int w  = (i >> 8) & 15;
    int h  = (i >> 12) & 15;
    int im = i >> 16;
    dst[i] = (bf16)src[(((size_t)im * 256 + c) * 16 + h) * 16 + w];
  }
}

// conv weight (Cout,Cin,3,3) -> (Cout,3,3,Cin) bf16  (== [N][K], K=(kh*3+kw)*Cin+c)
__global__ void pack_convw_k(const float* __restrict__ src, bf16* __restrict__ dst,
                             int Cout, int Cin) {
  int total = Cout * Cin * 9;
  for (int i = blockIdx.x * blockDim.x + threadIdx.x; i < total; i += gridDim.x * blockDim.x) {
    int c  = i % Cin;
    int r  = i / Cin;          // n*9 + kh*3 + kw
    int kw = r % 3;
    int kh = (r / 3) % 3;
    int n  = r / 9;
    dst[i] = (bf16)src[(((size_t)n * Cin + c) * 3 + kh) * 3 + kw];
  }
}

// fc1 weight (Nout,4096) with k_ref = c*16 + (h*4+w) -> bf16 [Nout][4096] with
// k_my = (h*4+w)*256 + c  (to match NHWC flatten of conv5 output)
__global__ void pack_fc1w_k(const float* __restrict__ src, bf16* __restrict__ dst, int Nout) {
  int total = Nout * 4096;
  for (int i = blockIdx.x * blockDim.x + threadIdx.x; i < total; i += gridDim.x * blockDim.x) {
    int kmy = i & 4095;
    int n   = i >> 12;
    int c   = kmy & 255;
    int hw  = kmy >> 8;
    dst[i] = (bf16)src[(size_t)n * 4096 + c * 16 + hw];
  }
}

__global__ void pack_cast_k(const float* __restrict__ src, bf16* __restrict__ dst, int total) {
  for (int i = blockIdx.x * blockDim.x + threadIdx.x; i < total; i += gridDim.x * blockDim.x)
    dst[i] = (bf16)src[i];
}

// ---------------------------------------------------------------------------
// Implicit-GEMM 3x3 conv (pad=1) + folded-BN affine + ReLU, bf16 in/out.
//   GEMM: M = IMGS*Hout*Wout, N = Cout, K = Cin*9.
//   Block tile 64x64, 8 wave32 waves (4 along M x 2 along N), K staged 64 wide
//   -> 4x v_wmma_f32_16x16x32_bf16 per LDS stage.
//   Loop order: kh, kw (tap decode + bounds hoisted) then c0 over channels.
// in  : [IMGS][Hin][Win][Cin]   bf16 (NHWC)
// wgt : [Cout][3][3][Cin]       bf16 ([N][K])
// out : [IMGS][Hout][Wout][Cout] bf16 (NHWC)
// ---------------------------------------------------------------------------
__global__ __launch_bounds__(256) void conv_bn_relu_wmma(
    const bf16* __restrict__ in, const bf16* __restrict__ wgt,
    const float* __restrict__ scale, const float* __restrict__ shift,
    bf16* __restrict__ out,
    int Cin, int Cout, int Hin, int Win, int Hout, int Wout, int stride) {
  __shared__ bf16 As[64][80];   // 64 pixels x 64 K (+16 pad, rows stay 16B aligned)
  __shared__ bf16 Bs[64][80];   // 64 couts  x 64 K

  const int t    = threadIdx.x;
  const int lane = t & 31;
  const int wid  = t >> 5;
  const int wm   = wid & 3;     // M subtile
  const int wn   = wid >> 2;    // N subtile (32 wide)
  const int Ktot = Cin * 9;

  // A loader: thread owns (row, 8-ch chunk); loads chunk and chunk+32 per stage
  const int rowA = t >> 2;
  const int chA  = (t & 3) << 3;
  const int mA   = blockIdx.x * 64 + rowA;
  const int owA  = mA % Wout;
  const int ohA  = (mA / Wout) % Hout;
  const int imgA = mA / (Wout * Hout);

  // B loader
  const int rowB = t >> 2;
  const int chB  = (t & 3) << 3;
  const int gnB  = blockIdx.y * 64 + rowB;
  const size_t wbase = (size_t)gnB * Ktot + chB;

  // fragment coordinates (loop-invariant)
  const int ar  = wm * 16 + (lane & 15);
  const int ak  = (lane >> 4) << 3;    // 0 or 8
  const int bk  = (lane >> 4) << 3;    // 0 or 8
  const int br0 = wn * 32 + (lane & 15);
  const int br1 = br0 + 16;

  v8f acc0 = vzero8f();
  v8f acc1 = vzero8f();

  for (int kh = 0; kh < 3; ++kh) {
    const int  ih  = ohA * stride + kh - 1;
    const bool okh = (unsigned)ih < (unsigned)Hin;
    for (int kw = 0; kw < 3; ++kw) {
      const int  iw = owA * stride + kw - 1;
      const bool ok = okh && ((unsigned)iw < (unsigned)Win);
      const bf16* abase = in + (((size_t)imgA * Hin + ih) * Win + iw) * Cin + chA;
      const size_t wtap = wbase + (size_t)(kh * 3 + kw) * Cin;

      for (int c0 = 0; c0 < Cin; c0 += 64) {
        // ---- stage A tile (zero-fill for padding) ----
        bf16x8 a0 = bzero8(), a1 = bzero8();
        if (ok) {
          a0 = *(const bf16x8*)(abase + c0);
          a1 = *(const bf16x8*)(abase + c0 + 32);
        }
        *(bf16x8*)&As[rowA][chA]      = a0;
        *(bf16x8*)&As[rowA][chA + 32] = a1;

        // ---- stage B tile (always in-bounds -> async global->LDS if avail) ----
#if HAVE_ASYNC_LDS
        async_b128(wgt + wtap + c0,      &Bs[rowB][chB]);
        async_b128(wgt + wtap + c0 + 32, &Bs[rowB][chB + 32]);
        async_wait0();
#else
        *(bf16x8*)&Bs[rowB][chB]      = *(const bf16x8*)(wgt + wtap + c0);
        *(bf16x8*)&Bs[rowB][chB + 32] = *(const bf16x8*)(wgt + wtap + c0 + 32);
#endif
        if (c0 + 64 < Cin)
          __builtin_prefetch(wgt + wtap + c0 + 64, 0, 1);  // global_prefetch_b8

        __syncthreads();

        // ---- 2 K-steps x 2 N-subtiles = 4 WMMA ----
#pragma unroll
        for (int ks = 0; ks < 2; ++ks) {
          const int o = ks << 5;
          v16bf afrag;
          ((bf16x8*)&afrag)[0] = *(const bf16x8*)&As[ar][o + ak];
          ((bf16x8*)&afrag)[1] = *(const bf16x8*)&As[ar][o + ak + 16];
          v16bf bf0, bf1;
          ((bf16x8*)&bf0)[0] = *(const bf16x8*)&Bs[br0][o + bk];
          ((bf16x8*)&bf0)[1] = *(const bf16x8*)&Bs[br0][o + bk + 16];
          ((bf16x8*)&bf1)[0] = *(const bf16x8*)&Bs[br1][o + bk];
          ((bf16x8*)&bf1)[1] = *(const bf16x8*)&Bs[br1][o + bk + 16];
          acc0 = __builtin_amdgcn_wmma_f32_16x16x32_bf16(false, afrag, false, bf0,
                                                         (short)0, acc0, false, false);
          acc1 = __builtin_amdgcn_wmma_f32_16x16x32_bf16(false, afrag, false, bf1,
                                                         (short)0, acc1, false, false);
        }
        __syncthreads();
      }
    }
  }

  // ---- epilogue: D layout -> (m = g + (lane>>4)*8, n = lane&15) ----
#pragma unroll
  for (int s = 0; s < 2; ++s) {
    v8f acc = s ? acc1 : acc0;
    const int ncol = blockIdx.y * 64 + wn * 32 + s * 16 + (lane & 15);
    const float sc = scale[ncol];
    const float sh = shift[ncol];
#pragma unroll
    for (int g = 0; g < 8; ++g) {
      const int mrow = blockIdx.x * 64 + wm * 16 + ((lane >> 4) << 3) + g;
      const int ow = mrow % Wout;
      const int oh = (mrow / Wout) % Hout;
      const int im = mrow / (Wout * Hout);
      float v = acc[g] * sc + sh;
      v = v > 0.0f ? v : 0.0f;
      out[(((size_t)im * Hout + oh) * Wout + ow) * Cout + ncol] = (bf16)v;
    }
  }
}

// ---------------------------------------------------------------------------
// Generic WMMA GEMM: out = act(A[M][K] * W[N][K]^T + bias), bf16 in, bf16/f32 out.
// Block tile 32x64, 8 waves (2 along M x 4 along N), K staged 64 wide
// -> 2x v_wmma per stage per wave.  K % 64 == 0, M % 32 == 0 assumed.
// ---------------------------------------------------------------------------
__global__ __launch_bounds__(256) void gemm_bias_act_wmma(
    const bf16* __restrict__ A, const bf16* __restrict__ W,
    const float* __restrict__ bias,
    int M, int N, int K, int do_relu,
    bf16* __restrict__ out_bf, float* __restrict__ out_f) {
  __shared__ bf16 As[32][80];
  __shared__ bf16 Bs[64][80];

  const int t    = threadIdx.x;
  const int lane = t & 31;
  const int wid  = t >> 5;
  const int wm   = wid & 1;
  const int wn   = wid >> 1;

  const int rowA = t >> 3;            // 0..31
  const int chA  = (t & 7) << 3;      // 16B chunks, one per thread per stage
  const size_t abase = (size_t)(blockIdx.x * 32 + rowA) * K + chA;

  const int rowB = t >> 2;            // 0..63
  const int chB  = (t & 3) << 3;      // two 16B chunks per thread per stage
  const int gnB  = blockIdx.y * 64 + rowB;
  const size_t wbase = (size_t)gnB * K + chB;

  const int ar = wm * 16 + (lane & 15);
  const int ak = (lane >> 4) << 3;
  const int br = wn * 16 + (lane & 15);
  const int bk = (lane >> 4) << 3;

  v8f acc = vzero8f();

  for (int k0 = 0; k0 < K; k0 += 64) {
    *(bf16x8*)&As[rowA][chA] = *(const bf16x8*)(A + abase + k0);
#if HAVE_ASYNC_LDS
    if (gnB < N) {
      async_b128(W + wbase + k0,      &Bs[rowB][chB]);
      async_b128(W + wbase + k0 + 32, &Bs[rowB][chB + 32]);
    } else {
      *(bf16x8*)&Bs[rowB][chB]      = bzero8();
      *(bf16x8*)&Bs[rowB][chB + 32] = bzero8();
    }
    async_wait0();
#else
    bf16x8 b0 = bzero8(), b1 = bzero8();
    if (gnB < N) {
      b0 = *(const bf16x8*)(W + wbase + k0);
      b1 = *(const bf16x8*)(W + wbase + k0 + 32);
    }
    *(bf16x8*)&Bs[rowB][chB]      = b0;
    *(bf16x8*)&Bs[rowB][chB + 32] = b1;
#endif
    if (gnB < N && k0 + 64 < K)
      __builtin_prefetch(W + wbase + k0 + 64, 0, 1);
    __syncthreads();

#pragma unroll
    for (int ks = 0; ks < 2; ++ks) {
      const int o = ks << 5;
      v16bf afrag;
      ((bf16x8*)&afrag)[0] = *(const bf16x8*)&As[ar][o + ak];
      ((bf16x8*)&afrag)[1] = *(const bf16x8*)&As[ar][o + ak + 16];
      v16bf bfrag;
      ((bf16x8*)&bfrag)[0] = *(const bf16x8*)&Bs[br][o + bk];
      ((bf16x8*)&bfrag)[1] = *(const bf16x8*)&Bs[br][o + bk + 16];
      acc = __builtin_amdgcn_wmma_f32_16x16x32_bf16(false, afrag, false, bfrag,
                                                    (short)0, acc, false, false);
    }
    __syncthreads();
  }

  const int ncol = blockIdx.y * 64 + wn * 16 + (lane & 15);
  if (ncol < N) {
    const float b = bias ? bias[ncol] : 0.0f;
#pragma unroll
    for (int g = 0; g < 8; ++g) {
      const int mrow = blockIdx.x * 32 + wm * 16 + ((lane >> 4) << 3) + g;
      float v = acc[g] + b;
      if (do_relu) v = v > 0.0f ? v : 0.0f;
      if (out_bf) out_bf[(size_t)mrow * N + ncol] = (bf16)v;
      if (out_f)  out_f[(size_t)mrow * N + ncol]  = v;
    }
  }
}

// ---------------------------------------------------------------------------
// Attention head (faithful to reference quirks): one block per batch b.
// ---------------------------------------------------------------------------
__global__ __launch_bounds__(256) void attention_k(
    const float* __restrict__ keys,   // [160][1024], row = i*32+b
    const float* __restrict__ qrys,   // [160][32],   row = j*32+b
    const float* __restrict__ aw,     // [1024][32]
    const float* __restrict__ ab,     // [1024]
    float* __restrict__ wsum) {       // [32][5]
  __shared__ float keyS[5][1024];
  __shared__ float qS[5][1024];
  __shared__ float qm[5][32];
  __shared__ float attn[25];
  const int b = blockIdx.x;
  const int t = threadIdx.x;

  for (int i = t; i < 160; i += 256) {
    int j = i >> 5, m = i & 31;
    qm[j][m] = qrys[(size_t)(j * 32 + b) * 32 + m];
  }
  for (int i = t; i < 5120; i += 256) {
    int r = i >> 10, n = i & 1023;
    keyS[r][n] = keys[(size_t)(r * 32 + b) * 1024 + n];
  }
  __syncthreads();
  for (int i = t; i < 5120; i += 256) {
    int j = i >> 10, n = i & 1023;
    float s = ab[n];
#pragma unroll 8
    for (int m = 0; m < 32; ++m) s += qm[j][m] * aw[n * 32 + m];
    qS[j][n] = s;
  }
  __syncthreads();
  if (t < 25) {
    int i = t / 5, j = t % 5;
    float s = 0.0f;
    for (int n = 0; n < 1024; ++n) s += keyS[i][n] * qS[j][n];
    attn[t] = s;
  }
  __syncthreads();
  if (t < 5) {
    int j = t;
    float d[4];
#pragma unroll
    for (int r = 0; r < 4; ++r) d[r] = (j > r) ? attn[r * 5 + j] : attn[(r + 1) * 5 + j];
    float mx = d[0];
#pragma unroll
    for (int r = 1; r < 4; ++r) mx = d[r] > mx ? d[r] : mx;
    float e[4], den = 0.0f;
#pragma unroll
    for (int r = 0; r < 4; ++r) { e[r] = __expf(d[r] - mx); den += e[r]; }
    float s = 0.0f;
#pragma unroll
    for (int r = 0; r < 4; ++r) s += e[r] / den;   // column sum of `final`
    wsum[b * 5 + j] = s;
  }
}

// out[b, :] = (1/5) * sum_q wsum[b][q] * bevs[b, q, :]   (f32)
__global__ void fuse_k(const float* __restrict__ bevs, const float* __restrict__ wsum,
                       float* __restrict__ out, int total) {
  for (int i = blockIdx.x * blockDim.x + threadIdx.x; i < total; i += gridDim.x * blockDim.x) {
    int b = i >> 16;           // C*H*W = 65536
    int o = i & 65535;
    const float* base = bevs + (size_t)b * 5 * 65536 + o;
    float s = 0.0f;
#pragma unroll
    for (int q = 0; q < 5; ++q) s += wsum[b * 5 + q] * base[(size_t)q * 65536];
    out[i] = 0.2f * s;
  }
}

// ---------------------------------------------------------------------------
// Host launcher
// ---------------------------------------------------------------------------
extern "C" void kernel_launch(void* const* d_in, const int* in_sizes, int n_in,
                              void* d_out, int out_size, void* d_ws, size_t ws_size,
                              hipStream_t stream) {
  (void)in_sizes; (void)n_in; (void)out_size; (void)ws_size;
  const float* bevs = (const float*)d_in[0];
  const float* cw[5]; const float* sN[5]; const float* tN[5];
  for (int i = 0; i < 5; ++i) {
    cw[i] = (const float*)d_in[1 + 3 * i];
    sN[i] = (const float*)d_in[2 + 3 * i];
    tN[i] = (const float*)d_in[3 + 3 * i];
  }
  const float* kw1 = (const float*)d_in[16]; const float* kb1 = (const float*)d_in[17];
  const float* kw2 = (const float*)d_in[18]; const float* kb2 = (const float*)d_in[19];
  const float* kw3 = (const float*)d_in[20]; const float* kb3 = (const float*)d_in[21];
  const float* qw1 = (const float*)d_in[22]; const float* qb1 = (const float*)d_in[23];
  const float* qw2 = (const float*)d_in[24]; const float* qb2 = (const float*)d_in[25];
  const float* qw3 = (const float*)d_in[26]; const float* qb3 = (const float*)d_in[27];
  const float* aw  = (const float*)d_in[28]; const float* ab  = (const float*)d_in[29];
  float* out = (float*)d_out;

  char* ws = (char*)d_ws;
  size_t off = 0;
  auto take = [&](size_t bytes) -> char* {
    char* p = ws + off;
    off = (off + bytes + 255) & ~(size_t)255;
    return p;
  };
  // Ping-pong activation buffers (bf16 NHWC)
  bf16* actA = (bf16*)take(160ull * 16 * 16 * 256 * 2);  // bevs/c2/c4
  bf16* actB = (bf16*)take(160ull * 16 * 16 * 512 * 2);  // c1/c3/c5 (feats)
  // Packed bf16 weights
  bf16* w1p  = (bf16*)take(512ull * 2304 * 2);
  bf16* w2p  = (bf16*)take(256ull * 4608 * 2);
  bf16* w3p  = (bf16*)take(256ull * 2304 * 2);
  bf16* w4p  = (bf16*)take(256ull * 2304 * 2);
  bf16* w5p  = (bf16*)take(256ull * 2304 * 2);
  bf16* kw1p = (bf16*)take(256ull * 4096 * 2);
  bf16* qw1p = (bf16*)take(256ull * 4096 * 2);
  bf16* kw2p = (bf16*)take(128ull * 256 * 2);
  bf16* qw2p = (bf16*)take(128ull * 256 * 2);
  bf16* kw3p = (bf16*)take(1024ull * 128 * 2);
  bf16* qw3p = (bf16*)take(32ull * 128 * 2);
  // MLP activations
  bf16*  h1k  = (bf16*)take(160ull * 256 * 2);
  bf16*  h1q  = (bf16*)take(160ull * 256 * 2);
  bf16*  h2k  = (bf16*)take(160ull * 128 * 2);
  bf16*  h2q  = (bf16*)take(160ull * 128 * 2);
  float* keys = (float*)take(160ull * 1024 * 4);
  float* qrys = (float*)take(160ull * 32 * 4);
  float* wsum = (float*)take(32ull * 5 * 4);

  // --- pack / convert ---
  pack_bevs_k<<<4096, 256, 0, stream>>>(bevs, actA, 160 * 16 * 16 * 256);
  pack_convw_k<<<1024, 256, 0, stream>>>(cw[0], w1p, 512, 256);
  pack_convw_k<<<1024, 256, 0, stream>>>(cw[1], w2p, 256, 512);
  pack_convw_k<<<512, 256, 0, stream>>>(cw[2], w3p, 256, 256);
  pack_convw_k<<<512, 256, 0, stream>>>(cw[3], w4p, 256, 256);
  pack_convw_k<<<512, 256, 0, stream>>>(cw[4], w5p, 256, 256);
  pack_fc1w_k<<<1024, 256, 0, stream>>>(kw1, kw1p, 256);
  pack_fc1w_k<<<1024, 256, 0, stream>>>(qw1, qw1p, 256);
  pack_cast_k<<<64, 256, 0, stream>>>(kw2, kw2p, 128 * 256);
  pack_cast_k<<<64, 256, 0, stream>>>(qw2, qw2p, 128 * 256);
  pack_cast_k<<<256, 256, 0, stream>>>(kw3, kw3p, 1024 * 128);
  pack_cast_k<<<16, 256, 0, stream>>>(qw3, qw3p, 32 * 128);

  // --- conv stack (implicit GEMM + WMMA), ping-pong actA <-> actB ---
  conv_bn_relu_wmma<<<dim3(640, 8), 256, 0, stream>>>(actA, w1p, sN[0], tN[0], actB,
                                                      256, 512, 16, 16, 16, 16, 1);
  conv_bn_relu_wmma<<<dim3(640, 4), 256, 0, stream>>>(actB, w2p, sN[1], tN[1], actA,
                                                      512, 256, 16, 16, 16, 16, 1);
  conv_bn_relu_wmma<<<dim3(160, 4), 256, 0, stream>>>(actA, w3p, sN[2], tN[2], actB,
                                                      256, 256, 16, 16, 8, 8, 2);
  conv_bn_relu_wmma<<<dim3(160, 4), 256, 0, stream>>>(actB, w4p, sN[3], tN[3], actA,
                                                      256, 256, 8, 8, 8, 8, 1);
  conv_bn_relu_wmma<<<dim3(40, 4), 256, 0, stream>>>(actA, w5p, sN[4], tN[4], actB,
                                                     256, 256, 8, 8, 4, 4, 2);

  // feats = actB viewed as [160][4096] bf16 (NHWC flatten; fc1 K is permuted to match)
  gemm_bias_act_wmma<<<dim3(5, 4),  256, 0, stream>>>(actB, kw1p, kb1, 160, 256, 4096, 1, h1k, nullptr);
  gemm_bias_act_wmma<<<dim3(5, 2),  256, 0, stream>>>(h1k,  kw2p, kb2, 160, 128, 256,  1, h2k, nullptr);
  gemm_bias_act_wmma<<<dim3(5, 16), 256, 0, stream>>>(h2k,  kw3p, kb3, 160, 1024, 128, 0, nullptr, keys);
  gemm_bias_act_wmma<<<dim3(5, 4),  256, 0, stream>>>(actB, qw1p, qb1, 160, 256, 4096, 1, h1q, nullptr);
  gemm_bias_act_wmma<<<dim3(5, 2),  256, 0, stream>>>(h1q,  qw2p, qb2, 160, 128, 256,  1, h2q, nullptr);
  gemm_bias_act_wmma<<<dim3(5, 1),  256, 0, stream>>>(h2q,  qw3p, qb3, 160, 32,  128,  0, nullptr, qrys);

  // --- attention + fuse ---
  attention_k<<<32, 256, 0, stream>>>(keys, qrys, aw, ab, wsum);
  fuse_k<<<4096, 256, 0, stream>>>(bevs, wsum, out, 32 * 256 * 16 * 16);
}
